// DSConv1d_75694503625198
// MI455X (gfx1250) — matrix operand order repaired and checked by reading, not compiled, not run on hardware
//
#include <hip/hip_runtime.h>

// ---------------------------------------------------------------------------
// MI455X (gfx1250) fused TCN block:
//  relu -> BN(eval) -> dwconv3 -> PReLU -> gLN -> 1x1 conv (GEMM) -> +x
//
// Element-wise ops fp32; the 512x512 channel GEMM in bf16 via
// V_WMMA_F32_16X16X32_BF16 (fp32 accum). gLN folded into GEMM weights +
// epilogue. h staged in workspace as bf16 [B][T_pad][C] (T_pad=4096, pad
// rows zeroed) so GEMM staging is unconditional GLOBAL_LOAD_ASYNC_TO_LDS_B128
// into double-buffered LDS, overlapping HBM/L2 latency with WMMA issue.
// ---------------------------------------------------------------------------

typedef __attribute__((ext_vector_type(16))) __bf16 v16bf;
typedef __attribute__((ext_vector_type(8)))  __bf16 v8bf;
typedef __attribute__((ext_vector_type(8)))  float  v8f;

#define CC 512
#define TT 4000
#define TP 4096            // padded T for h_t so GEMM tiles never go OOB
#define BB 16
#define NBLK_PER_B 1008    // 63 t-tiles * 16 c-tiles in kernel 1

__device__ __forceinline__ unsigned short f2bf(float f) {
    unsigned u = __builtin_bit_cast(unsigned, f);
    u += 0x7FFFu + ((u >> 16) & 1u);   // round-to-nearest-even bf16
    return (unsigned short)(u >> 16);
}

// ---------------------------------------------------------------------------
// Kernel 1: relu + BN + depthwise conv(3, pad 1) + PReLU
//   -> h_t (bf16, [B][TP][C]) + per-block partial (sum, sumsq) for gLN
// ---------------------------------------------------------------------------
__global__ __launch_bounds__(256)
void k_fused_pre(const float* __restrict__ x,
                 const float* __restrict__ bng, const float* __restrict__ bnb,
                 const float* __restrict__ bnm, const float* __restrict__ bnv,
                 const float* __restrict__ dw,  const float* __restrict__ pa,
                 unsigned short* __restrict__ ht, float* __restrict__ partials)
{
    __shared__ float xs[32][66];
    __shared__ float bnsc[32], bnsh[32];
    __shared__ float w0s[32], w1s[32], w2s[32];
    __shared__ unsigned short hs[32 * 64];
    __shared__ float red[256];

    const int tid = threadIdx.x;
    const int t0 = blockIdx.x * 64;
    const int c0 = blockIdx.y * 32;
    const int b  = blockIdx.z;

    if (tid < 32) {
        int c = c0 + tid;
        float inv = rsqrtf(bnv[c] + 1e-5f);
        float sc  = bng[c] * inv;
        bnsc[tid] = sc;
        bnsh[tid] = bnb[c] - bnm[c] * sc;
        w0s[tid]  = dw[c * 3 + 0];
        w1s[tid]  = dw[c * 3 + 1];
        w2s[tid]  = dw[c * 3 + 2];
    }
    __syncthreads();

    for (int e = tid; e < 32 * 66; e += 256) {
        int cl = e / 66, tl = e % 66;
        int gt = t0 + tl - 1;
        float v = 0.0f;
        if (gt >= 0 && gt < TT) {
            float xv = x[((size_t)b * CC + (c0 + cl)) * TT + gt];
            v = fmaxf(xv, 0.0f) * bnsc[cl] + bnsh[cl];
        }
        xs[cl][tl] = v;
    }
    __syncthreads();

    const float alpha = pa[0];
    float lsum = 0.0f, lsq = 0.0f;
#pragma unroll
    for (int it = 0; it < 8; ++it) {
        int e = tid + it * 256;
        int cl = e >> 6, tl = e & 63;
        float h = w0s[cl] * xs[cl][tl] + w1s[cl] * xs[cl][tl + 1]
                + w2s[cl] * xs[cl][tl + 2];
        h = (h >= 0.0f) ? h : alpha * h;
        if (t0 + tl < TT) { lsum += h; lsq += h * h; } else { h = 0.0f; }
        hs[cl * 64 + tl] = f2bf(h);
    }
    __syncthreads();

    for (int e = tid; e < 32 * 64; e += 256) {
        int cl = e & 31, tl = e >> 5;
        int gt = t0 + tl;
        if (gt < TT)
            ht[((size_t)b * TP + gt) * CC + c0 + cl] = hs[cl * 64 + tl];
    }

    red[tid] = lsum; __syncthreads();
    for (int s = 128; s > 0; s >>= 1) { if (tid < s) red[tid] += red[tid + s]; __syncthreads(); }
    float bsum = red[0]; __syncthreads();
    red[tid] = lsq; __syncthreads();
    for (int s = 128; s > 0; s >>= 1) { if (tid < s) red[tid] += red[tid + s]; __syncthreads(); }
    if (tid == 0) {
        int pid = blockIdx.y * 63 + blockIdx.x;
        partials[((size_t)b * NBLK_PER_B + pid) * 2 + 0] = bsum;
        partials[((size_t)b * NBLK_PER_B + pid) * 2 + 1] = red[0];
    }
}

// ---------------------------------------------------------------------------
// Kernel 1b: zero the pad rows t in [TT, TP) of h_t  (one block per batch)
// ---------------------------------------------------------------------------
__global__ __launch_bounds__(256)
void k_pad_zero(unsigned short* __restrict__ ht)
{
    const int b = blockIdx.x;
    uint4* p = (uint4*)(ht + ((size_t)b * TP + TT) * CC);
    const int n = (TP - TT) * CC / 8;      // uint4 count = 6144
    uint4 z = make_uint4(0u, 0u, 0u, 0u);
    for (int i = threadIdx.x; i < n; i += 256) p[i] = z;
}

// ---------------------------------------------------------------------------
// Kernel 2: per-batch mean / rstd
// ---------------------------------------------------------------------------
__global__ __launch_bounds__(256)
void k_stats(const float* __restrict__ partials, float* __restrict__ meanrs)
{
    __shared__ float rs_[256], rq_[256];
    const int b = blockIdx.x, tid = threadIdx.x;
    float s = 0.0f, q = 0.0f;
    for (int i = tid; i < NBLK_PER_B; i += 256) {
        s += partials[((size_t)b * NBLK_PER_B + i) * 2 + 0];
        q += partials[((size_t)b * NBLK_PER_B + i) * 2 + 1];
    }
    rs_[tid] = s; rq_[tid] = q; __syncthreads();
    for (int st = 128; st > 0; st >>= 1) {
        if (tid < st) { rs_[tid] += rs_[tid + st]; rq_[tid] += rq_[tid + st]; }
        __syncthreads();
    }
    if (tid == 0) {
        const float invN = 1.0f / (float)(CC * TT);
        float mean = rs_[0] * invN;
        float var  = rq_[0] * invN - mean * mean;
        meanrs[b * 2 + 0] = mean;
        meanrs[b * 2 + 1] = rsqrtf(var + 1e-8f);
    }
}

// ---------------------------------------------------------------------------
// Kernel 3: fold gLN gamma into weights (bf16) + row sums for bias terms
// ---------------------------------------------------------------------------
__global__ __launch_bounds__(256)
void k_fold(const float* __restrict__ pw, const float* __restrict__ g,
            const float* __restrict__ gb, unsigned short* __restrict__ wf,
            float* __restrict__ rowg, float* __restrict__ rowb)
{
    __shared__ float r1[256], r2[256];
    const int o = blockIdx.x, tid = threadIdx.x;
    float s1 = 0.0f, s2 = 0.0f;
    for (int c = tid; c < CC; c += 256) {
        float w0 = pw[o * CC + c];
        float w  = w0 * g[c];
        wf[o * CC + c] = f2bf(w);
        s1 += w;
        s2 += w0 * gb[c];
    }
    r1[tid] = s1; r2[tid] = s2; __syncthreads();
    for (int st = 128; st > 0; st >>= 1) {
        if (tid < st) { r1[tid] += r1[tid + st]; r2[tid] += r2[tid + st]; }
        __syncthreads();
    }
    if (tid == 0) { rowg[o] = r1[0]; rowb[o] = r2[0]; }
}

// ---------------------------------------------------------------------------
// Kernel 4: per-batch GEMM  out[o][t] = rs*( W' x h ) + bias[o] + x[o][t]
// 128x128 block tile, 8 waves * (32x64) = 2x4 wmma accums each, K-step 32.
// Staging: GLOBAL_LOAD_ASYNC_TO_LDS_B128 into double-buffered LDS; prefetch
// of tile k+1 overlaps the WMMAs of tile k; s_wait_asynccnt + barrier flips.
// ---------------------------------------------------------------------------
__global__ __launch_bounds__(256)
void k_gemm(const unsigned short* __restrict__ wf,
            const unsigned short* __restrict__ ht,
            const float* __restrict__ rowg, const float* __restrict__ rowb,
            const float* __restrict__ meanrs,
            const float* __restrict__ x, float* __restrict__ out)
{
    __shared__ __align__(16) unsigned short a_lds[2][128 * 32]; // [m][k]
    __shared__ __align__(16) unsigned short b_lds[2][128 * 32]; // [n][k]

    const int tid   = threadIdx.x;
    const int lane  = tid & 31;
    const int wave  = tid >> 5;
    const int wm    = wave & 3;
    const int wn    = wave >> 2;
    const int lm    = lane & 15;
    const int lh    = lane >> 4;

    const int t0      = blockIdx.x * 128;
    const int m_block = blockIdx.y * 128;
    const int b       = blockIdx.z;

    v8f acc[2][4];
#pragma unroll
    for (int i = 0; i < 2; ++i)
#pragma unroll
        for (int j = 0; j < 4; ++j) acc[i][j] = (v8f)(0.0f);

    // per-thread staging geometry: 2 chunks each for A and B per K-step
    const int ch0  = tid;            // 0..255
    const int ch1  = tid + 256;      // 256..511
    const int row0 = ch0 >> 2, kc0 = (ch0 & 3) * 8;
    const int row1 = ch1 >> 2, kc1 = (ch1 & 3) * 8;

    // async DMA: global (b128) -> LDS, tracked by ASYNCcnt (no VGPR round-trip)
    auto stage = [&](int k0, int p) {
        {
            unsigned la = (unsigned)(size_t)&a_lds[p][row0 * 32 + kc0];
            const unsigned short* ga = wf + (size_t)(m_block + row0) * CC + k0 + kc0;
            asm volatile("global_load_async_to_lds_b128 %0, %1, off"
                         :: "v"(la), "v"(ga) : "memory");
            unsigned lb = (unsigned)(size_t)&b_lds[p][row0 * 32 + kc0];
            const unsigned short* gb2 = ht + ((size_t)b * TP + (t0 + row0)) * CC + k0 + kc0;
            asm volatile("global_load_async_to_lds_b128 %0, %1, off"
                         :: "v"(lb), "v"(gb2) : "memory");
        }
        {
            unsigned la = (unsigned)(size_t)&a_lds[p][row1 * 32 + kc1];
            const unsigned short* ga = wf + (size_t)(m_block + row1) * CC + k0 + kc1;
            asm volatile("global_load_async_to_lds_b128 %0, %1, off"
                         :: "v"(la), "v"(ga) : "memory");
            unsigned lb = (unsigned)(size_t)&b_lds[p][row1 * 32 + kc1];
            const unsigned short* gb2 = ht + ((size_t)b * TP + (t0 + row1)) * CC + k0 + kc1;
            asm volatile("global_load_async_to_lds_b128 %0, %1, off"
                         :: "v"(lb), "v"(gb2) : "memory");
        }
    };

    stage(0, 0);
    asm volatile("s_wait_asynccnt 0x0" ::: "memory");
    __syncthreads();

    int p = 0;
    for (int k0 = 0; k0 < CC; k0 += 32) {
        if (k0 + 32 < CC) stage(k0 + 32, p ^ 1);   // prefetch next tile

        // A fragments: lane holds row m, contiguous K run of 16 at lh*16
        v16bf af[2];
#pragma unroll
        for (int sm = 0; sm < 2; ++sm) {
            int m = wm * 32 + sm * 16 + lm;
            const v8bf* ap = (const v8bf*)&a_lds[p][0];
            v8bf lo = ap[m * 4 + 2 * lh];
            v8bf hi = ap[m * 4 + 2 * lh + 1];
            af[sm] = __builtin_shufflevector(lo, hi,
                     0,1,2,3,4,5,6,7,8,9,10,11,12,13,14,15);
        }
        // B fragments: lane holds col n, contiguous K run of 16 at lh*16
        v16bf bfr[4];
#pragma unroll
        for (int sn = 0; sn < 4; ++sn) {
            int n = wn * 64 + sn * 16 + lm;
            const v16bf* bp = (const v16bf*)&b_lds[p][0];
            bfr[sn] = bp[n * 2 + lh];
        }

#pragma unroll
        for (int sm = 0; sm < 2; ++sm)
#pragma unroll
            for (int sn = 0; sn < 4; ++sn)
                acc[sm][sn] = __builtin_amdgcn_wmma_f32_16x16x32_bf16(
                    false, af[sm], false, bfr[sn],
                    (short)0, acc[sm][sn], false, false);

        asm volatile("s_wait_asynccnt 0x0" ::: "memory");
        __syncthreads();
        p ^= 1;
    }

    const float mean = meanrs[b * 2 + 0];
    const float rs   = meanrs[b * 2 + 1];
#pragma unroll
    for (int sm = 0; sm < 2; ++sm) {
#pragma unroll
        for (int r = 0; r < 8; ++r) {
            int o = m_block + wm * 32 + sm * 16 + lh * 8 + r;
            float bias = rowb[o] - rs * mean * rowg[o];
#pragma unroll
            for (int sn = 0; sn < 4; ++sn) {
                int t = t0 + wn * 64 + sn * 16 + lm;
                if (t < TT) {
                    size_t idx = ((size_t)b * CC + o) * TT + t;
                    out[idx] = acc[sm][sn][r] * rs + bias + x[idx];
                }
            }
        }
    }
}

// ---------------------------------------------------------------------------
// launcher -- workspace layout (bytes), total ~64.6 MB:
//   [0)           h_t bf16  B*TP*C*2 = 67,108,864
//   [67,108,864)  W'  bf16  512*512*2 =   524,288
//   [67,633,152)  rowg f32  512*4     =     2,048
//   [67,635,200)  rowb f32  512*4     =     2,048
//   [67,637,248)  partials  B*1008*2*4=   129,024
//   [67,766,272)  meanrs    B*2*4     =       128
// ---------------------------------------------------------------------------
extern "C" void kernel_launch(void* const* d_in, const int* in_sizes, int n_in,
                              void* d_out, int out_size, void* d_ws, size_t ws_size,
                              hipStream_t stream)
{
    const float* x    = (const float*)d_in[0];
    const float* bng  = (const float*)d_in[1];
    const float* bnb  = (const float*)d_in[2];
    const float* bnm  = (const float*)d_in[3];
    const float* bnv  = (const float*)d_in[4];
    const float* dw   = (const float*)d_in[5];
    const float* pa   = (const float*)d_in[6];
    const float* gg   = (const float*)d_in[7];
    const float* gb   = (const float*)d_in[8];
    const float* pw   = (const float*)d_in[9];
    float* out = (float*)d_out;

    char* ws = (char*)d_ws;
    unsigned short* ht   = (unsigned short*)(ws);
    unsigned short* wf   = (unsigned short*)(ws + 67108864);
    float* rowg          = (float*)(ws + 67633152);
    float* rowb          = (float*)(ws + 67635200);
    float* partials      = (float*)(ws + 67637248);
    float* meanrs        = (float*)(ws + 67766272);

    k_fused_pre<<<dim3(63, 16, BB), 256, 0, stream>>>(
        x, bng, bnb, bnm, bnv, dw, pa, ht, partials);
    k_pad_zero<<<dim3(BB), 256, 0, stream>>>(ht);
    k_stats<<<dim3(BB), 256, 0, stream>>>(partials, meanrs);
    k_fold<<<dim3(CC), 256, 0, stream>>>(pw, gg, gb, wf, rowg, rowb);
    k_gemm<<<dim3(32, 4, BB), 256, 0, stream>>>(
        wf, ht, rowg, rowb, meanrs, x, out);
}
